// Net_27986006901491
// MI455X (gfx1250) — matrix operand order repaired and checked by reading, not compiled, not run on hardware
//
#include <hip/hip_runtime.h>

// ---------------------------------------------------------------------------
// MI455X (gfx1250) implementation.
// Bound by streaming 5x 64MB fp32 weight matrices from HBM (~13.7us floor at
// 23.3 TB/s). Compute requirement (~0.6 TFLOP/s) is far below the WMMA
// ceiling, so we spend free FLOPs on a 2-term bf16 split (hi+lo) of both
// operands and issue 3x v_wmma_f32_16x16x32_bf16 per tile to get ~fp32-grade
// accuracy while using the CDNA5 matrix path. wave32 throughout.
// ---------------------------------------------------------------------------

typedef __attribute__((ext_vector_type(16))) __bf16 v16bf;
typedef __attribute__((ext_vector_type(8)))  __bf16 v8bf;
typedef __attribute__((ext_vector_type(8)))  float  v8f;

#define FDIM   4096
#define NEDGE  44
#define NNODE  14
#define MPAD   48   // 3 WMMA M-tiles of 16

// ---------------------------------------------------------------------------
// Graph propagation (linear operator on the 44xF edge matrix), fused with the
// fp32 -> (bf16 hi, bf16 lo) split of the GEMM A operand. One thread per
// feature column; per-thread node scratch lives in LDS (column-per-thread, so
// no barriers and no bank conflicts: bank = tid % 64).
// ---------------------------------------------------------------------------
__global__ __launch_bounds__(256)
void graphprep_kernel(const float* __restrict__ E, const int* __restrict__ esrc,
                      const int* __restrict__ edst, float* __restrict__ eh,
                      __bf16* __restrict__ Ahi, __bf16* __restrict__ Alo)
{
    const int t = threadIdx.x;
    const int f = blockIdx.x * 256 + t;

    __shared__ float nodeS [NNODE][256];
    __shared__ float node2S[NNODE][256];
    __shared__ float degS  [NNODE][256];

    #pragma unroll
    for (int n = 0; n < NNODE; ++n) { nodeS[n][t] = 0.f; node2S[n][t] = 0.f; degS[n][t] = 0.f; }

    // node = segment_sum(edge_h by dst) / deg
    for (int e = 0; e < NEDGE; ++e) {
        const int d = edst[e];
        nodeS[d][t] += E[e * FDIM + f];
        degS [d][t] += 1.0f;
    }
    #pragma unroll
    for (int n = 0; n < NNODE; ++n) nodeS[n][t] /= degS[n][t];

    // node2 = segment_sum(node[src] by dst)
    for (int e = 0; e < NEDGE; ++e) node2S[edst[e]][t] += nodeS[esrc[e]][t];

    // eh = 0.5 * (node2[src] + node2[dst]); emit f32 + bf16 hi/lo split
    for (int e = 0; e < NEDGE; ++e) {
        const float v = 0.5f * (node2S[esrc[e]][t] + node2S[edst[e]][t]);
        eh[e * FDIM + f] = v;
        const __bf16 h = (__bf16)v;
        Ahi[e * FDIM + f] = h;
        Alo[e * FDIM + f] = (__bf16)(v - (float)h);
    }
    // zero the padded rows so WMMA on M-tile 2 contributes nothing there
    for (int e = NEDGE; e < MPAD; ++e) {
        Ahi[e * FDIM + f] = (__bf16)0.f;
        Alo[e * FDIM + f] = (__bf16)0.f;
    }
}

// Assemble the 16-bit A-matrix lane layout (ISA 7.12.2):
// lanes 0-15 : M = lane,    K = [0..7] then [16..23]
// lanes 16-31: M = lane-16, K = [8..15] then [24..31]
__device__ __forceinline__ v16bf load_a16(const __bf16* __restrict__ row, int kb)
{
    const v8bf c0 = *(const v8bf*)(row + kb);
    const v8bf c1 = *(const v8bf*)(row + kb + 16);
    v16bf r;
    #pragma unroll
    for (int i = 0; i < 8; ++i) { r[i] = c0[i]; r[i + 8] = c1[i]; }
    return r;
}

// ---------------------------------------------------------------------------
// out[m, n0:n0+16] = eh[m, :] @ W[n, :]^T + b, optional leaky-ReLU.
// One block per N-tile of 16 (256 blocks). 8 waves per block, each owning a
// disjoint K-chunk of 512 and all 3 M-tiles (weight reuse + amortized f32->
// bf16 split of B). K-partials reduced through LDS; bias+act fused.
// ---------------------------------------------------------------------------
__global__ __launch_bounds__(256, 1)
void gemm48_kernel(const __bf16* __restrict__ Ahi, const __bf16* __restrict__ Alo,
                   const float* __restrict__ W, const float* __restrict__ bias,
                   float* __restrict__ out, int do_act)
{
    const int tid  = threadIdx.x;
    const int wave = tid >> 5;
    const int lane = tid & 31;
    const int n0   = blockIdx.x * 16;

    const int lh = lane >> 4;       // half-wave select
    const int ml = lane & 15;       // M within tile
    const int kbase = wave * (FDIM / 8);            // 512-wide K chunk
    const int klane = ml + (lh << 4);               // B row (K) owned by lane

    const __bf16* a0h = Ahi + (0 * 16 + ml) * FDIM;
    const __bf16* a1h = Ahi + (1 * 16 + ml) * FDIM;
    const __bf16* a2h = Ahi + (2 * 16 + ml) * FDIM;
    const __bf16* a0l = Alo + (0 * 16 + ml) * FDIM;
    const __bf16* a1l = Alo + (1 * 16 + ml) * FDIM;
    const __bf16* a2l = Alo + (2 * 16 + ml) * FDIM;

    v8f acc0 = {}, acc1 = {}, acc2 = {};

    #pragma unroll 2
    for (int ks = 0; ks < FDIM / 8; ks += 32) {
        const int k0 = kbase + ks;
        const int kb = k0 + (lh << 3);

        // A tiles (pre-split bf16, streamed from L2)
        const v16bf A0h = load_a16(a0h, kb), A0l = load_a16(a0l, kb);
        const v16bf A1h = load_a16(a1h, kb), A1l = load_a16(a1l, kb);
        const v16bf A2h = load_a16(a2h, kb), A2l = load_a16(a2l, kb);

        // B tile: B[k, n] = W[n0+n, k]. Lane holds column N = lane%16,
        // elems = K [8h..8h+7] then [16+8h..16+8h+7] (same as A).
        const float* wp = W + (size_t)(n0 + ml) * FDIM + (k0 + (lh << 3));
        v16bf Bh, Bl;
        #pragma unroll
        for (int n = 0; n < 16; ++n) {
            const float f = wp[(n & 7) + ((n >> 3) << 4)];
            const __bf16 h = (__bf16)f;
            Bh[n] = h;
            Bl[n] = (__bf16)(f - (float)h);
        }

        // two-term split product: Ah*Bh + Ah*Bl + Al*Bh (Al*Bl ~ 2^-16, dropped)
        acc0 = __builtin_amdgcn_wmma_f32_16x16x32_bf16(false, A0h, false, Bh, (short)0, acc0, false, false);
        acc0 = __builtin_amdgcn_wmma_f32_16x16x32_bf16(false, A0h, false, Bl, (short)0, acc0, false, false);
        acc0 = __builtin_amdgcn_wmma_f32_16x16x32_bf16(false, A0l, false, Bh, (short)0, acc0, false, false);
        acc1 = __builtin_amdgcn_wmma_f32_16x16x32_bf16(false, A1h, false, Bh, (short)0, acc1, false, false);
        acc1 = __builtin_amdgcn_wmma_f32_16x16x32_bf16(false, A1h, false, Bl, (short)0, acc1, false, false);
        acc1 = __builtin_amdgcn_wmma_f32_16x16x32_bf16(false, A1l, false, Bh, (short)0, acc1, false, false);
        acc2 = __builtin_amdgcn_wmma_f32_16x16x32_bf16(false, A2h, false, Bh, (short)0, acc2, false, false);
        acc2 = __builtin_amdgcn_wmma_f32_16x16x32_bf16(false, A2h, false, Bl, (short)0, acc2, false, false);
        acc2 = __builtin_amdgcn_wmma_f32_16x16x32_bf16(false, A2l, false, Bh, (short)0, acc2, false, false);
    }

    // cross-wave K reduction through LDS, fused bias + leaky-ReLU epilogue.
    __shared__ float red[8][3][16][16];
    // C/D layout (ISA 7.12.2): VGPR r, lanes 0-15 -> (M=r, N=lane);
    // lanes 16-31 -> (M=r+8, N=lane-16).
    #pragma unroll
    for (int r = 0; r < 8; ++r) {
        red[wave][0][r + lh * 8][ml] = acc0[r];
        red[wave][1][r + lh * 8][ml] = acc1[r];
        red[wave][2][r + lh * 8][ml] = acc2[r];
    }
    __syncthreads();

    const int m = tid >> 4;
    const int n = tid & 15;
    #pragma unroll
    for (int mt = 0; mt < 3; ++mt) {
        float s = 0.f;
        #pragma unroll
        for (int w2 = 0; w2 < 8; ++w2) s += red[w2][mt][m][n];
        s += bias[n0 + n];
        if (do_act) s = (s > 0.f) ? s : 0.01f * s;
        const int mg = mt * 16 + m;
        if (mg < NEDGE) out[mg * FDIM + n0 + n] = s;
    }
}

// ---------------------------------------------------------------------------
// h2 = eh @ Wv5^T + bv5  (44 x 4096 @ 4096 x 1) — tiny, plain reduction.
// ---------------------------------------------------------------------------
__global__ __launch_bounds__(256)
void v5_kernel(const float* __restrict__ eh, const float* __restrict__ W,
               const float* __restrict__ b, float* __restrict__ out)
{
    const int e = blockIdx.x;
    const int t = threadIdx.x;
    float s = 0.f;
    for (int k = t; k < FDIM; k += 256) s += eh[e * FDIM + k] * W[k];
    __shared__ float red[256];
    red[t] = s;
    __syncthreads();
    for (int off = 128; off > 0; off >>= 1) {
        if (t < off) red[t] += red[t + off];
        __syncthreads();
    }
    if (t == 0) out[e] = red[0] + b[0];
}

extern "C" void kernel_launch(void* const* d_in, const int* in_sizes, int n_in,
                              void* d_out, int out_size, void* d_ws, size_t ws_size,
                              hipStream_t stream)
{
    (void)in_sizes; (void)n_in; (void)out_size; (void)ws_size;

    const float* x    = (const float*)d_in[0];
    const int*   esrc = (const int*)  d_in[1];
    const int*   edst = (const int*)  d_in[2];
    const float* Wa1 = (const float*)d_in[3];  const float* ba1 = (const float*)d_in[4];
    const float* Wa2 = (const float*)d_in[5];  const float* ba2 = (const float*)d_in[6];
    const float* Wa5 = (const float*)d_in[7];  const float* ba5 = (const float*)d_in[8];
    const float* Wv1 = (const float*)d_in[9];  const float* bv1 = (const float*)d_in[10];
    const float* Wv2 = (const float*)d_in[11]; const float* bv2 = (const float*)d_in[12];
    const float* Wv5 = (const float*)d_in[13]; const float* bv5 = (const float*)d_in[14];

    float* h1 = (float*)d_out;                  // 44 x 4096
    float* h2 = (float*)d_out + NEDGE * FDIM;   // 44

    // workspace layout (16B aligned regions), ~2.3 MB total
    char* ws = (char*)d_ws;
    float*  eh   = (float*)(ws);                                  // 48*4096 f32
    __bf16* Ahi  = (__bf16*)(ws + (size_t)MPAD * FDIM * 4);       // 48*4096 bf16
    __bf16* Alo  = (__bf16*)(ws + (size_t)MPAD * FDIM * 4 + (size_t)MPAD * FDIM * 2);
    float*  hbuf = (float*)(ws + (size_t)MPAD * FDIM * 4 + (size_t)MPAD * FDIM * 4);

    const dim3 gp(FDIM / 256), bp(256);   // graph-prep: one thread per feature
    const dim3 gg(FDIM / 16),  bg(256);   // gemm: one block per N-tile of 16

    // a-branch
    graphprep_kernel<<<gp, bp, 0, stream>>>(x,    esrc, edst, eh, Ahi, Alo);
    gemm48_kernel  <<<gg, bg, 0, stream>>>(Ahi, Alo, Wa1, ba1, hbuf, 1);
    graphprep_kernel<<<gp, bp, 0, stream>>>(hbuf, esrc, edst, eh, Ahi, Alo);
    gemm48_kernel  <<<gg, bg, 0, stream>>>(Ahi, Alo, Wa2, ba2, hbuf, 1);
    graphprep_kernel<<<gp, bp, 0, stream>>>(hbuf, esrc, edst, eh, Ahi, Alo);
    gemm48_kernel  <<<gg, bg, 0, stream>>>(Ahi, Alo, Wa5, ba5, h1, 0);

    // v-branch
    graphprep_kernel<<<gp, bp, 0, stream>>>(x,    esrc, edst, eh, Ahi, Alo);
    gemm48_kernel  <<<gg, bg, 0, stream>>>(Ahi, Alo, Wv1, bv1, hbuf, 1);
    graphprep_kernel<<<gp, bp, 0, stream>>>(hbuf, esrc, edst, eh, Ahi, Alo);
    gemm48_kernel  <<<gg, bg, 0, stream>>>(Ahi, Alo, Wv2, bv2, hbuf, 1);
    graphprep_kernel<<<gp, bp, 0, stream>>>(hbuf, esrc, edst, eh, Ahi, Alo);
    v5_kernel      <<<dim3(NEDGE), dim3(256), 0, stream>>>(eh, Wv5, bv5, h2);
}
